// LocalMaskedMHCA_20444044329309
// MI455X (gfx1250) — matrix operand order repaired and checked by reading, not compiled, not run on hardware
//
#include <hip/hip_runtime.h>
#include <hip/hip_bf16.h>

#define BB 4
#define CC 512
#define TT 2304
#define HH 16
#define DD 32
#define BT (BB*TT)            // 9216 positions
#define EPSLN 1e-5f
#define SCALE 0.17677669529663687f   // 1/sqrt(32)

typedef __bf16 v16bf __attribute__((ext_vector_type(16)));
typedef float  v8f   __attribute__((ext_vector_type(8)));
typedef unsigned int u32x4 __attribute__((ext_vector_type(4)));
typedef int          i32x4 __attribute__((ext_vector_type(4)));
typedef int          i32x8 __attribute__((ext_vector_type(8)));

union AB  { v16bf v; unsigned u[8]; unsigned short h[16]; };
union ACC { v8f v; float f[8]; };

__device__ __forceinline__ unsigned short f2bf(float f) {
  unsigned u = __float_as_uint(f);
  u += 0x7FFFu + ((u >> 16) & 1u);        // round-to-nearest-even
  return (unsigned short)(u >> 16);
}
__device__ __forceinline__ v8f zero8() {
  v8f z = {0.f,0.f,0.f,0.f,0.f,0.f,0.f,0.f};
  return z;
}
// A-matrix (16x32 bf16) K index of VGPR-pair p for this half-wave (ISA 7.12.2)
__device__ __forceinline__ int amapK(int p, int hi) {
  return (p < 4) ? ((hi ? 8 : 0) + 2*p) : ((hi ? 24 : 16) + 2*(p - 4));
}

// ---------------------------------------------------------------------------
// TDM: async 2-D bf16 tile load global->LDS (D# per CDNA5 ISA).
// Tile rows are 32 bf16 (=16 DWORDs); pad 4 DWORDs per row -> LDS row stride
// of 40 bf16 elements, matching the padded shared arrays below.
// ---------------------------------------------------------------------------
__device__ __forceinline__ void tdm_load_2d(unsigned lds_off,
                                            const unsigned short* gtile,
                                            unsigned tensor_rows,
                                            unsigned row_len_elems,
                                            unsigned tile_cols,
                                            unsigned tile_rows)
{
  unsigned long long ga = (unsigned long long)(size_t)gtile;
  u32x4 g0;
  g0.x = 0x1u;                                        // count=1, user D#
  g0.y = lds_off;                                     // LDS byte address
  g0.z = (unsigned)(ga & 0xFFFFFFFFu);                // global_addr[31:0]
  g0.w = (unsigned)((ga >> 32) & 0x1FFFFFFu)          // global_addr[56:32]
       | (2u << 30);                                  // type=2 ("image")
  i32x8 g1;
  g1[0] = (1 << 16)                                   // data_size = 2 bytes
        | (1 << 20)                                   // pad_enable
        | (3 << 22)                                   // pad_interval: 16 DW
        | (3 << 25);                                  // pad_amount: 4 DW
  g1[1] = (int)((row_len_elems & 0xFFFFu) << 16);     // tensor_dim0[15:0]
  g1[2] = (int)((row_len_elems >> 16)                 // tensor_dim0[31:16]
        | ((tensor_rows & 0xFFFFu) << 16));           // tensor_dim1[15:0]
  g1[3] = (int)((tensor_rows >> 16)                   // tensor_dim1[31:16]
        | (tile_cols << 16));                         // tile_dim0
  g1[4] = (int)tile_rows;                             // tile_dim1 (tile_dim2=0)
  g1[5] = (int)row_len_elems;                         // tensor_dim0_stride lo
  g1[6] = 0;
  g1[7] = 0;
  i32x4 z4 = {0, 0, 0, 0};
#if defined(__clang_major__) && (__clang_major__ >= 23)
  i32x8 z8 = {0, 0, 0, 0, 0, 0, 0, 0};
  __builtin_amdgcn_tensor_load_to_lds(g0, g1, z4, z4, z8, 0);
#else
  __builtin_amdgcn_tensor_load_to_lds(g0, g1, z4, z4, 0);
#endif
}

// ---------------------------------------------------------------------------
// Weight f32 -> bf16 pre-conversion (once per launch; 1 MB per matrix)
// ---------------------------------------------------------------------------
__global__ __launch_bounds__(256)
void w2bf_kernel(const float* __restrict__ w, unsigned short* __restrict__ o, int n)
{
  int i = blockIdx.x*256 + threadIdx.x;
  if (i < n) o[i] = f2bf(w[i]);
}

// ---------------------------------------------------------------------------
// Stage 1: depthwise conv3 (+mask) + LayerNorm over C, emit bf16 act [n][C]
// ---------------------------------------------------------------------------
__global__ __launch_bounds__(256)
void prep_ln_kernel(const float* __restrict__ x, const float* __restrict__ wc,
                    const float* __restrict__ g, const float* __restrict__ be,
                    const unsigned char* __restrict__ mask,
                    unsigned short* __restrict__ act)
{
  __shared__ float ssum[256], ssq[256];
  int n = blockIdx.x;
  int b = n / TT, t = n % TT;
  int tid = threadIdx.x;
  float mf = mask[n] ? 1.f : 0.f;
  float y[2];
#pragma unroll
  for (int i = 0; i < 2; ++i) {
    int c = tid + i*256;
    const float* xr = x + ((size_t)b*CC + c)*TT;
    float xm = (t > 0)    ? xr[t-1] : 0.f;
    float x0 = xr[t];
    float xp = (t < TT-1) ? xr[t+1] : 0.f;
    y[i] = (xm*wc[c*3+0] + x0*wc[c*3+1] + xp*wc[c*3+2]) * mf;
  }
  ssum[tid] = y[0] + y[1];
  ssq[tid]  = y[0]*y[0] + y[1]*y[1];
  __syncthreads();
  for (int s = 128; s > 0; s >>= 1) {
    if (tid < s) { ssum[tid] += ssum[tid+s]; ssq[tid] += ssq[tid+s]; }
    __syncthreads();
  }
  float mean = ssum[0] * (1.f/CC);
  float var  = ssq[0]  * (1.f/CC) - mean*mean;
  float rs = rsqrtf(var + EPSLN);
#pragma unroll
  for (int i = 0; i < 2; ++i) {
    int c = tid + i*256;
    act[(size_t)n*CC + c] = f2bf((y[i] - mean) * rs * g[c] + be[c]);
  }
}

// ---------------------------------------------------------------------------
// Stage 2/5: WMMA GEMM  D[o,n] = sum_c Wbf[o,c]*act[n,c] + bias[o]
// TDM double-buffered: wave 0 issues two tensor_load_to_lds per K-chunk,
// pipelined one chunk ahead, synced with s_wait_tensorcnt.
// Block tile 128(M) x 64(N), 8 waves each owning 32x32, K-chunks of 32.
// ---------------------------------------------------------------------------
template<int MODE>
__global__ __launch_bounds__(256)
void gemm_wmma_kernel(const unsigned short* __restrict__ Wbf,
                      const float* __restrict__ bias,
                      const unsigned short* __restrict__ act,
                      unsigned short* __restrict__ outh, float* __restrict__ outf,
                      const unsigned char* __restrict__ mask)
{
  __shared__ unsigned short sW[2][128][40];   // [buf][m][k], TDM-padded rows
  __shared__ unsigned short sA[2][64][40];    // [buf][n][k]
  int tid  = threadIdx.x;
  int lane = tid & 31, wid = tid >> 5;
  int hi   = (lane >= 16);
  int m0 = blockIdx.y * 128;
  int n0 = blockIdx.x * 64;
  int wm = (wid & 3) * 32;
  int wn = (wid >> 2) * 32;

  v8f acc[2][2];
#pragma unroll
  for (int i = 0; i < 2; ++i)
#pragma unroll
    for (int j = 0; j < 2; ++j) acc[i][j] = zero8();

  // prologue: chunk 0 -> buffer 0
  if (wid == 0) {
    tdm_load_2d((unsigned)(size_t)&sW[0][0][0], Wbf + (size_t)m0*CC,
                (unsigned)(CC - m0), CC, 32, 128);
    tdm_load_2d((unsigned)(size_t)&sA[0][0][0], act + (size_t)n0*CC,
                (unsigned)(BT - n0), CC, 32, 64);
  }

  for (int it = 0; it < CC/32; ++it) {
    int cur = it & 1;
    if (wid == 0) {
      if (it + 1 < CC/32) {
        int kc = (it + 1) * 32;
        tdm_load_2d((unsigned)(size_t)&sW[cur^1][0][0],
                    Wbf + (size_t)m0*CC + kc, (unsigned)(CC - m0), CC, 32, 128);
        tdm_load_2d((unsigned)(size_t)&sA[cur^1][0][0],
                    act + (size_t)n0*CC + kc, (unsigned)(BT - n0), CC, 32, 64);
        __builtin_amdgcn_s_wait_tensorcnt((short)2);  // chunk `it` landed
      } else {
        __builtin_amdgcn_s_wait_tensorcnt((short)0);
      }
    }
    __syncthreads();                                  // LDS visible to all waves

    AB a[2], bfr[2];
#pragma unroll
    for (int i = 0; i < 2; ++i) {                     // A fragments (W rows)
      int row = wm + i*16 + (lane & 15);
#pragma unroll
      for (int p = 0; p < 8; ++p)
        a[i].u[p] = *(const unsigned*)&sW[cur][row][amapK(p, hi)];
    }
#pragma unroll
    for (int j = 0; j < 2; ++j) {                     // B fragments (act cols)
      int row = wn + j*16 + (lane & 15);
      int kb  = hi ? 16 : 0;
#pragma unroll
      for (int p = 0; p < 8; ++p)
        bfr[j].u[p] = *(const unsigned*)&sA[cur][row][amapK(p, hi)];
    }
#pragma unroll
    for (int i = 0; i < 2; ++i)
#pragma unroll
      for (int j = 0; j < 2; ++j)
        acc[i][j] = __builtin_amdgcn_wmma_f32_16x16x32_bf16(
            false, a[i].v, false, bfr[j].v, (short)0, acc[i][j], false, false);
    __syncthreads();                                  // gate buffer reuse
  }

#pragma unroll
  for (int i = 0; i < 2; ++i) {
#pragma unroll
    for (int j = 0; j < 2; ++j) {
      ACC av; av.v = acc[i][j];
      int nn = n0 + wn + j*16 + (lane & 15);
      int bb = nn / TT, tt = nn % TT;
#pragma unroll
      for (int r = 0; r < 8; ++r) {
        int oo = m0 + wm + i*16 + r + (hi ? 8 : 0);
        float val = av.f[r] + bias[oo];
        if (MODE == 0) {
          int hd = oo >> 5, d = oo & 31;
          outh[(((size_t)bb*HH + hd)*TT + tt)*DD + d] = f2bf(val);
        } else {
          float mf = mask[nn] ? 1.f : 0.f;
          outf[((size_t)bb*CC + oo)*TT + tt] = val * mf;
        }
      }
    }
  }
}

// ---------------------------------------------------------------------------
// Stage 3: banded attention. One wave per (b, h, 16-row q tile).
// Keys covered by 48-wide strip [t0-16, t0+31] ⊇ band [t0-9, t0+24].
// 3 WMMAs for scores, lane-shuffle softmax, 4 WMMAs for probs@V.
// ---------------------------------------------------------------------------
__global__ __launch_bounds__(32)
void attn_wmma_kernel(const unsigned short* __restrict__ qh,
                      const unsigned short* __restrict__ kh,
                      const unsigned short* __restrict__ vh,
                      const unsigned char* __restrict__ mask,
                      unsigned short* __restrict__ ao)
{
  __shared__ float sS[16][52];
  int lane = threadIdx.x & 31;
  int hi   = lane >= 16;
  int t0 = blockIdx.x * 16;
  int h  = blockIdx.y;
  int b  = blockIdx.z;
  int kb = t0 - 16;
  const unsigned short* qbase = qh + (((size_t)b*HH + h)*TT)*DD;
  const unsigned short* kbase = kh + (((size_t)b*HH + h)*TT)*DD;
  const unsigned short* vbase = vh + (((size_t)b*HH + h)*TT)*DD;

  AB qa;                                         // Q A-fragment: 16 t-rows x 32 d
  {
    int t = t0 + (lane & 15);
    const unsigned short* qr = qbase + (size_t)t*DD;
#pragma unroll
    for (int p = 0; p < 8; ++p)
      qa.u[p] = *(const unsigned*)(qr + amapK(p, hi));
  }

  ACC sc[3];                                     // scores: q @ K^T per 16-key tile
#pragma unroll
  for (int jt = 0; jt < 3; ++jt) {
    AB kf;
    int pk = kb + jt*16 + (lane & 15);
    int db = hi ? 16 : 0;
    if (pk >= 0 && pk < TT) {
      const unsigned short* kr = kbase + (size_t)pk*DD;
#pragma unroll
      for (int p = 0; p < 8; ++p) kf.u[p] = *(const unsigned*)(kr + amapK(p, hi));
    } else {
#pragma unroll
      for (int p = 0; p < 8; ++p) kf.u[p] = 0u;
    }
    sc[jt].v = __builtin_amdgcn_wmma_f32_16x16x32_bf16(
        false, qa.v, false, kf.v, (short)0, zero8(), false, false);
  }

  // band mask + additive key mask + row softmax (16-lane shfl_xor reductions)
#pragma unroll
  for (int r = 0; r < 8; ++r) {
    int tloc = r + (hi ? 8 : 0);
    float m = -3.0e30f;
#pragma unroll
    for (int jt = 0; jt < 3; ++jt) {
      int j  = jt*16 + (lane & 15);
      int pk = kb + j;
      int dlt = j - tloc;                        // key - query + 16 ∈ [7,25]
      bool band = (dlt >= 7) && (dlt <= 25) && (pk >= 0) && (pk < TT);
      float s = sc[jt].f[r] * SCALE;
      if (band) { if (!mask[b*TT + pk]) s -= 1.0e4f; }
      else        s = -1.0e30f;
      sc[jt].f[r] = s;
      m = fmaxf(m, s);
    }
#pragma unroll
    for (int off = 1; off < 16; off <<= 1) m = fmaxf(m, __shfl_xor(m, off, 32));
    float sum = 0.f;
#pragma unroll
    for (int jt = 0; jt < 3; ++jt) {
      float e = __expf(sc[jt].f[r] - m);
      sc[jt].f[r] = e;
      sum += e;
    }
#pragma unroll
    for (int off = 1; off < 16; off <<= 1) sum += __shfl_xor(sum, off, 32);
    float inv = 1.f / (sum + 1e-20f);
#pragma unroll
    for (int jt = 0; jt < 3; ++jt) sc[jt].f[r] *= inv;
  }

  // C-layout probs -> LDS -> A-layout bf16 fragments
#pragma unroll
  for (int r = 0; r < 8; ++r) {
    int row = r + (hi ? 8 : 0);
#pragma unroll
    for (int jt = 0; jt < 3; ++jt)
      sS[row][jt*16 + (lane & 15)] = sc[jt].f[r];
  }
  __syncthreads();

  v8f oacc[2] = { zero8(), zero8() };            // out = P(16x48→64) @ V(48x32)
#pragma unroll
  for (int c = 0; c < 2; ++c) {
    AB pa;
    int row = lane & 15;
#pragma unroll
    for (int p = 0; p < 8; ++p) {
      int K = c*32 + (hi ? 16 : 0) + 2*p;
      float f0 = (K     < 48) ? sS[row][K]     : 0.f;
      float f1 = (K + 1 < 48) ? sS[row][K + 1] : 0.f;
      pa.h[2*p]   = f2bf(f0);
      pa.h[2*p+1] = f2bf(f1);
    }
#pragma unroll
    for (int nt = 0; nt < 2; ++nt) {
      AB vf;
      int d = nt*16 + (lane & 15);
#pragma unroll
      for (int j = 0; j < 16; ++j) {
        int K  = c*32 + (hi ? 16 : 0) + j;
        int pk = kb + K;
        vf.h[j] = (K < 48 && pk >= 0 && pk < TT)
                    ? vbase[(size_t)pk*DD + d] : (unsigned short)0;
      }
      oacc[nt] = __builtin_amdgcn_wmma_f32_16x16x32_bf16(
          false, pa.v, false, vf.v, (short)0, oacc[nt], false, false);
    }
  }

#pragma unroll
  for (int nt = 0; nt < 2; ++nt) {               // emit act-layout bf16 [n][C]
    ACC av; av.v = oacc[nt];
    int d = nt*16 + (lane & 15);
#pragma unroll
    for (int r = 0; r < 8; ++r) {
      int t = t0 + r + (hi ? 8 : 0);
      unsigned short val = mask[b*TT + t] ? f2bf(av.f[r]) : (unsigned short)0;
      ao[((size_t)b*TT + t)*CC + h*DD + d] = val;
    }
  }
}

// second tuple output: mask as float
__global__ void mask_out_kernel(const unsigned char* __restrict__ mask,
                                float* __restrict__ outm)
{
  int i = blockIdx.x*256 + threadIdx.x;
  if (i < BT) outm[i] = mask[i] ? 1.f : 0.f;
}

// ---------------------------------------------------------------------------
extern "C" void kernel_launch(void* const* d_in, const int* in_sizes, int n_in,
                              void* d_out, int out_size, void* d_ws, size_t ws_size,
                              hipStream_t stream)
{
  const float* x       = (const float*)d_in[0];
  const float* w_qc    = (const float*)d_in[1];
  const float* w_kc    = (const float*)d_in[2];
  const float* w_vc    = (const float*)d_in[3];
  const float* g_q     = (const float*)d_in[4];
  const float* b_q     = (const float*)d_in[5];
  const float* g_k     = (const float*)d_in[6];
  const float* b_k     = (const float*)d_in[7];
  const float* g_v     = (const float*)d_in[8];
  const float* b_v     = (const float*)d_in[9];
  const float* w_query = (const float*)d_in[10];
  const float* b_query = (const float*)d_in[11];
  const float* w_key   = (const float*)d_in[12];
  const float* b_key   = (const float*)d_in[13];
  const float* w_value = (const float*)d_in[14];
  const float* b_value = (const float*)d_in[15];
  const float* w_proj  = (const float*)d_in[16];
  const float* b_proj  = (const float*)d_in[17];
  const unsigned char* mask = (const unsigned char*)d_in[18];
  float* out = (float*)d_out;

  const size_t SZ = (size_t)CC * BT;             // bf16 elements per tensor
  const int    WN = CC * CC;                     // weight elements
  unsigned short* actq = (unsigned short*)d_ws;
  unsigned short* actk = actq + SZ;
  unsigned short* actv = actk + SZ;
  unsigned short* qhp  = actv + SZ;
  unsigned short* khp  = qhp + SZ;
  unsigned short* vhp  = khp + SZ;
  unsigned short* aop  = vhp + SZ;
  unsigned short* wqb  = aop + SZ;
  unsigned short* wkb  = wqb + WN;
  unsigned short* wvb  = wkb + WN;
  unsigned short* wpb  = wvb + WN;

  // weights f32 -> bf16 (once per launch)
  w2bf_kernel<<<(WN + 255)/256, 256, 0, stream>>>(w_query, wqb, WN);
  w2bf_kernel<<<(WN + 255)/256, 256, 0, stream>>>(w_key,   wkb, WN);
  w2bf_kernel<<<(WN + 255)/256, 256, 0, stream>>>(w_value, wvb, WN);
  w2bf_kernel<<<(WN + 255)/256, 256, 0, stream>>>(w_proj,  wpb, WN);

  prep_ln_kernel<<<BT, 256, 0, stream>>>(x, w_qc, g_q, b_q, mask, actq);
  prep_ln_kernel<<<BT, 256, 0, stream>>>(x, w_kc, g_k, b_k, mask, actk);
  prep_ln_kernel<<<BT, 256, 0, stream>>>(x, w_vc, g_v, b_v, mask, actv);

  dim3 gg(BT/64, CC/128);
  gemm_wmma_kernel<0><<<gg, 256, 0, stream>>>(wqb, b_query, actq, qhp, nullptr, mask);
  gemm_wmma_kernel<0><<<gg, 256, 0, stream>>>(wkb, b_key,   actk, khp, nullptr, mask);
  gemm_wmma_kernel<0><<<gg, 256, 0, stream>>>(wvb, b_value, actv, vhp, nullptr, mask);

  attn_wmma_kernel<<<dim3(TT/16, HH, BB), 32, 0, stream>>>(qhp, khp, vhp, mask, aop);

  gemm_wmma_kernel<1><<<gg, 256, 0, stream>>>(wpb, b_proj, aop, nullptr, out, mask);

  mask_out_kernel<<<(BT + 255)/256, 256, 0, stream>>>(mask, out + (size_t)BB*CC*TT);
}
